// QuantLinearCH_28810640621897
// MI455X (gfx1250) — 8-bit matrix operand order repaired and checked by reading, compiled, not run on hardware
//
#include <hip/hip_runtime.h>
#include <stdint.h>

typedef __attribute__((ext_vector_type(8))) int v8i;

#define BM     8192
#define CHN    4096
#define NOUT   4096
#define NGRP   16
#define PERG   256
#define KSTEPS (CHN / 64)
#define MTILES (BM / 16)
#define NTILES (NOUT / 16)

static constexpr size_t OFF_APACK = 0;
static constexpr size_t OFF_BPACK = 33554432;
static constexpr size_t OFF_MEAN  = 50331648;
static constexpr size_t OFF_HALF  = OFF_MEAN  + 16384;
static constexpr size_t OFF_CORR  = OFF_HALF  + 16384;
static constexpr size_t OFF_ASC   = OFF_CORR  + 16384;
static constexpr size_t OFF_IASC  = OFF_ASC   + 256;
static constexpr size_t OFF_WSC   = OFF_IASC  + 256;
static constexpr size_t OFF_SIDX  = OFF_WSC   + 256;
static constexpr size_t OFF_CMAXK = OFF_SIDX  + 16384;
static constexpr size_t OFF_CMINK = OFF_CMAXK + 16384;
static constexpr size_t OFF_WMAXK = OFF_CMINK + 16384;

__device__ __forceinline__ unsigned fkey(float f) {
  unsigned u = __float_as_uint(f);
  return (u & 0x80000000u) ? ~u : (u | 0x80000000u);
}
__device__ __forceinline__ float fdec(unsigned k) {
  return __uint_as_float((k & 0x80000000u) ? (k & 0x7FFFFFFFu) : ~k);
}
__device__ __forceinline__ int q8(float f) {
  int q = (int)rintf(f);
  return q < -128 ? -128 : (q > 127 ? 127 : q);
}

__global__ void k_init(unsigned* cmaxk, unsigned* cmink, unsigned* wmaxk) {
  int i = blockIdx.x * blockDim.x + threadIdx.x;
  if (i < CHN) { cmaxk[i] = 0u; cmink[i] = 0xFFFFFFFFu; }
  if (i == 0) wmaxk[0] = 0u;
}

__global__ void k_wmax(const float* __restrict__ w, unsigned* wmaxk) {
  __shared__ unsigned sm[256];
  float mx = -3.4e38f;
  for (size_t i = blockIdx.x * 256 + threadIdx.x; i < (size_t)NOUT * CHN; i += 256 * 256)
    mx = fmaxf(mx, w[i]);
  sm[threadIdx.x] = fkey(mx);
  __syncthreads();
  for (int s = 128; s > 0; s >>= 1) {
    if (threadIdx.x < s) sm[threadIdx.x] = max(sm[threadIdx.x], sm[threadIdx.x + s]);
    __syncthreads();
  }
  if (threadIdx.x == 0) atomicMax(wmaxk, sm[0]);
}

__global__ void k_chminmax(const float* __restrict__ x, unsigned* cmaxk, unsigned* cmink) {
  int c = blockIdx.x * 256 + threadIdx.x;
  int r0 = blockIdx.y * (BM / 64);
  float mx = -3.4e38f, mn = 3.4e38f;
  for (int r = 0; r < BM / 64; ++r) {
    float v = x[(size_t)(r0 + r) * CHN + c];
    mx = fmaxf(mx, v); mn = fminf(mn, v);
  }
  atomicMax(&cmaxk[c], fkey(mx));
  atomicMin(&cmink[c], fkey(mn));
}

__global__ void k_stats(const unsigned* cmaxk, const unsigned* cmink, const unsigned* wmaxk,
                        float* meanc, float* halfc, float* wsc) {
  int c = blockIdx.x * 256 + threadIdx.x;
  float mx = fdec(cmaxk[c]), mn = fdec(cmink[c]);
  meanc[c] = 0.5f * (mx + mn);
  halfc[c] = 0.5f * (mx - mn);
  if (c == 0) {
    float wm = fdec(wmaxk[0]);
    wsc[0] = wm / 127.0f;
    wsc[1] = 127.0f / wm;
  }
}

__global__ void k_rank(const float* __restrict__ halfc, int* sidx) {
  int i = blockIdx.x * 256 + threadIdx.x;
  float ai = halfc[i];
  int cnt = 0;
  for (int j = 0; j < CHN; ++j) {
    float aj = halfc[j];
    cnt += (aj < ai) || (aj == ai && j < i);
  }
  sidx[cnt] = i;
}

__global__ void k_ascale(const float* __restrict__ halfc, const int* __restrict__ sidx,
                         float* asc, float* iasc) {
  __shared__ float sm[256];
  int g = blockIdx.x;
  sm[threadIdx.x] = halfc[sidx[g * PERG + threadIdx.x]];
  __syncthreads();
  for (int s = 128; s > 0; s >>= 1) {
    if (threadIdx.x < s) sm[threadIdx.x] = fmaxf(sm[threadIdx.x], sm[threadIdx.x + s]);
    __syncthreads();
  }
  if (threadIdx.x == 0) {
    float m = sm[0];
    asc[g]  = m / 127.0f;
    iasc[g] = m > 0.0f ? 127.0f / m : 0.0f;
  }
}

__global__ void k_corr(const float* __restrict__ w, const float* __restrict__ bias,
                       const float* __restrict__ meanc, const float* __restrict__ wsc,
                       float* corr) {
  int n = blockIdx.x * 256 + threadIdx.x;
  float invw = wsc[1], acc = 0.0f;
  const float* wr = w + (size_t)n * CHN;
  for (int c = 0; c < CHN; ++c)
    acc += meanc[c] * (float)q8(wr[c] * invw);
  corr[n] = bias[n] + wsc[0] * acc;
}

__global__ void k_packX(const float* __restrict__ x, const float* __restrict__ meanc,
                        const int* __restrict__ sidx, const float* __restrict__ iasc,
                        int* __restrict__ Apack) {
  int kstep = blockIdx.x, mtile = blockIdx.y;
  int lane = threadIdx.x >> 3, v = threadIdx.x & 7;
  int m = mtile * 16 + (lane & 15);
  int kA = ((v >> 1) << 4) + ((lane >> 4) << 3) + ((v & 1) << 2);
  int kbase = kstep * 64 + kA;
  float ia = iasc[kbase >> 8];
  const float* xr = x + (size_t)m * CHN;
  unsigned dw = 0;
#pragma unroll
  for (int j = 0; j < 4; ++j) {
    int c = sidx[kbase + j];
    int q = q8((xr[c] - meanc[c]) * ia);
    dw |= (unsigned)(q & 255) << (8 * j);
  }
  Apack[((size_t)(mtile * KSTEPS + kstep) * 32 + lane) * 8 + v] = (int)dw;
}

__global__ void k_packW(const float* __restrict__ w, const int* __restrict__ sidx,
                        const float* __restrict__ wsc, int* __restrict__ Bpack) {
  int kstep = blockIdx.x, ntile = blockIdx.y;
  int lane = threadIdx.x >> 3, v = threadIdx.x & 7;
  int n = ntile * 16 + (lane & 15);
  int kB = ((v >> 1) << 4) + ((lane >> 4) << 3) + ((v & 1) << 2);
  int kbase = kstep * 64 + kB;
  float invw = wsc[1];
  const float* wr = w + (size_t)n * CHN;
  unsigned dw = 0;
#pragma unroll
  for (int j = 0; j < 4; ++j) {
    int c = sidx[kbase + j];
    int q = q8(wr[c] * invw);
    dw |= (unsigned)(q & 255) << (8 * j);
  }
  Bpack[((size_t)(ntile * KSTEPS + kstep) * 32 + lane) * 8 + v] = (int)dw;
}

__global__ void __launch_bounds__(256)
k_gemm(const int* __restrict__ Apack, const int* __restrict__ Bpack,
       const float* __restrict__ asc, const float* __restrict__ wsc,
       const float* __restrict__ corr, float* __restrict__ out) {
  const int lane = threadIdx.x & 31;
  const int wave = threadIdx.x >> 5;
  const int m0 = blockIdx.x * 64 + (wave & 1) * 32;
  const int n0 = blockIdx.y * 256 + (wave >> 1) * 64;

  const int* Ab[2];
  const int* Bb[4];
#pragma unroll
  for (int t = 0; t < 2; ++t)
    Ab[t] = Apack + ((size_t)((m0 >> 4) + t) * KSTEPS * 32 + lane) * 8;
#pragma unroll
  for (int t = 0; t < 4; ++t)
    Bb[t] = Bpack + ((size_t)((n0 >> 4) + t) * KSTEPS * 32 + lane) * 8;

  float facc[2][4][8];
#pragma unroll
  for (int i = 0; i < 2; ++i)
#pragma unroll
    for (int j = 0; j < 4; ++j)
#pragma unroll
      for (int e = 0; e < 8; ++e) facc[i][j][e] = 0.0f;

  for (int g = 0; g < NGRP; ++g) {
    v8i ia[2][4];
#pragma unroll
    for (int i = 0; i < 2; ++i)
#pragma unroll
      for (int j = 0; j < 4; ++j)
#pragma unroll
        for (int e = 0; e < 8; ++e) ia[i][j][e] = 0;

#pragma unroll
    for (int kk = 0; kk < 4; ++kk) {
      const int ko = (g * 4 + kk) * 256;
      v8i a[2], b[4];
#pragma unroll
      for (int t = 0; t < 2; ++t) a[t] = *(const v8i*)(Ab[t] + ko);
#pragma unroll
      for (int t = 0; t < 4; ++t) b[t] = *(const v8i*)(Bb[t] + ko);
#pragma unroll
      for (int tn = 0; tn < 4; ++tn)
#pragma unroll
        for (int tm = 0; tm < 2; ++tm)
          ia[tm][tn] = __builtin_amdgcn_wmma_i32_16x16x64_iu8(
              true, a[tm], true, b[tn], ia[tm][tn], false, false);
    }
    const float as = asc[g];
#pragma unroll
    for (int i = 0; i < 2; ++i)
#pragma unroll
      for (int j = 0; j < 4; ++j)
#pragma unroll
        for (int e = 0; e < 8; ++e) facc[i][j][e] += as * (float)ia[i][j][e];
  }

  const float w_s = wsc[0];
#pragma unroll
  for (int tm = 0; tm < 2; ++tm)
#pragma unroll
    for (int tn = 0; tn < 4; ++tn)
#pragma unroll
      for (int e = 0; e < 8; ++e) {
        int m = m0 + tm * 16 + ((lane >> 4) << 3) + e;
        int n = n0 + tn * 16 + (lane & 15);
        out[(size_t)m * NOUT + n] = w_s * facc[tm][tn][e] + corr[n];
      }
}

extern "C" void kernel_launch(void* const* d_in, const int* in_sizes, int n_in,
                              void* d_out, int out_size, void* d_ws, size_t ws_size,
                              hipStream_t stream) {
  const float* x    = (const float*)d_in[0];
  const float* w    = (const float*)d_in[1];
  const float* bias = (const float*)d_in[2];
  float* out = (float*)d_out;

  char* ws = (char*)d_ws;
  int*      Apack = (int*)(ws + OFF_APACK);
  int*      Bpack = (int*)(ws + OFF_BPACK);
  float*    meanc = (float*)(ws + OFF_MEAN);
  float*    halfc = (float*)(ws + OFF_HALF);
  float*    corr  = (float*)(ws + OFF_CORR);
  float*    asc   = (float*)(ws + OFF_ASC);
  float*    iasc  = (float*)(ws + OFF_IASC);
  float*    wsc   = (float*)(ws + OFF_WSC);
  int*      sidx  = (int*)(ws + OFF_SIDX);
  unsigned* cmaxk = (unsigned*)(ws + OFF_CMAXK);
  unsigned* cmink = (unsigned*)(ws + OFF_CMINK);
  unsigned* wmaxk = (unsigned*)(ws + OFF_WMAXK);

  k_init    <<<dim3(16),        256, 0, stream>>>(cmaxk, cmink, wmaxk);
  k_wmax    <<<dim3(256),       256, 0, stream>>>(w, wmaxk);
  k_chminmax<<<dim3(16, 64),    256, 0, stream>>>(x, cmaxk, cmink);
  k_stats   <<<dim3(16),        256, 0, stream>>>(cmaxk, cmink, wmaxk, meanc, halfc, wsc);
  k_rank    <<<dim3(16),        256, 0, stream>>>(halfc, sidx);
  k_ascale  <<<dim3(16),        256, 0, stream>>>(halfc, sidx, asc, iasc);
  k_corr    <<<dim3(16),        256, 0, stream>>>(w, bias, meanc, wsc, corr);
  k_packX   <<<dim3(KSTEPS, MTILES), 256, 0, stream>>>(x, meanc, sidx, iasc, Apack);
  k_packW   <<<dim3(KSTEPS, NTILES), 256, 0, stream>>>(w, sidx, wsc, Bpack);
  k_gemm    <<<dim3(BM / 64, NOUT / 256), 256, 0, stream>>>(Apack, Bpack, asc, wsc, corr, out);
}
